// MultiHeadAttention_59923383714456
// MI455X (gfx1250) — matrix operand order repaired and checked by reading, not compiled, not run on hardware
//
#include <hip/hip_runtime.h>

typedef __bf16 bf16_t;
typedef __attribute__((ext_vector_type(16))) __bf16 v16bf;
typedef __attribute__((ext_vector_type(8)))  __bf16 v8bf;
typedef __attribute__((ext_vector_type(8)))  float  v8f;
typedef __attribute__((ext_vector_type(4)))  float  v4f;
typedef __attribute__((ext_vector_type(2)))  unsigned v2u;
typedef __attribute__((ext_vector_type(4)))  unsigned v4u;
typedef __attribute__((ext_vector_type(8)))  unsigned v8u;

#define B_ 2
#define C_ 2048
#define M_ 1024
#define H_ 16

// ---- f32 -> bf16 (round-half-up) ------------------------------------------
// Pack two floats' bf16 forms into one dword: 2x v_add + 1x v_perm_b32.
__device__ __forceinline__ unsigned pack2bf(float a, float b) {
#if __has_builtin(__builtin_amdgcn_cvt_pk_bf16_f32)
  typedef __attribute__((ext_vector_type(2))) __bf16 v2bf;
  v2bf p = __builtin_amdgcn_cvt_pk_bf16_f32(a, b);
  return __builtin_bit_cast(unsigned, p);
#else
  unsigned ua = __builtin_bit_cast(unsigned, a) + 0x8000u;
  unsigned ub = __builtin_bit_cast(unsigned, b) + 0x8000u;
  // D.b16[0] = ua.hi16, D.b16[1] = ub.hi16
  return __builtin_amdgcn_perm(ub, ua, 0x07060302u);
#endif
}

// Scalar store: 1x v_add, then the compiler selects a *_store_b16_d16_hi.
__device__ __forceinline__ void store_bf16(bf16_t* p, float f) {
  unsigned u = __builtin_bit_cast(unsigned, f) + 0x8000u;
  *(unsigned short*)p = (unsigned short)(u >> 16);
}

__device__ __forceinline__ bf16_t f2bf(float f) {
  unsigned u = __builtin_bit_cast(unsigned, f) + 0x8000u;
  union { unsigned short s; bf16_t b; } y;
  y.s = (unsigned short)(u >> 16);
  return y.b;
}

__device__ __forceinline__ v8f wmma_bf16(v16bf a, v16bf b, v8f c) {
  return __builtin_amdgcn_wmma_f32_16x16x32_bf16(false, a, false, b, (short)0, c,
                                                 false, false);
}

__device__ __forceinline__ v16bf combine8(v4u lo, v4u hi) {
  v8u t;
  t[0] = lo[0]; t[1] = lo[1]; t[2] = lo[2]; t[3] = lo[3];
  t[4] = hi[0]; t[5] = hi[1]; t[6] = hi[2]; t[7] = hi[3];
  return __builtin_bit_cast(v16bf, t);
}

// A fragment 16x32 (M x K), src row-major src[m*ld + k]; two 16B runs per lane.
__device__ __forceinline__ v16bf load_A16(const bf16_t* src, int ld, int lane) {
  const int m  = lane & 15;
  const int kb = (lane & 16) ? 8 : 0;
  const bf16_t* p = src + m * ld + kb;
  v4u lo = *(const v4u*)p;
  v4u hi = *(const v4u*)(p + 16);
  return combine8(lo, hi);
}

// B fragment 32x16 (K x N) from a TRANSPOSED tile srcT[n*ld + k]; contiguous K.
__device__ __forceinline__ v16bf load_B16(const bf16_t* srcT, int ld, int lane) {
  const int nn = lane & 15;
  const int kb = (lane & 16) ? 8 : 0;
  const bf16_t* p = srcT + nn * ld + kb;
  v4u lo = *(const v4u*)p;
  v4u hi = *(const v4u*)(p + 16);
  return combine8(lo, hi);
}

// ---------------------------------------------------------------------------
// Kernel 1: QKV projection.  grid = (C/128, B*H, 3 [q/k/v]), block = 256.
// 1/sqrt(64) softmax scale folded into Q here.
// ---------------------------------------------------------------------------
__global__ __launch_bounds__(256) void qkv_kernel(
    const float* __restrict__ kvinput, const float* __restrict__ qinput,
    const float* __restrict__ wq, const float* __restrict__ wk,
    const float* __restrict__ wv,
    bf16_t* __restrict__ Qb, bf16_t* __restrict__ Kb, bf16_t* __restrict__ Vb) {
  const int which = blockIdx.z;
  const int bh = blockIdx.y, b = bh / H_, h = bh % H_;
  const int ct = blockIdx.x;
  const float* x = (which == 0) ? qinput : kvinput;
  const float* w = (which == 0) ? wq : (which == 1) ? wk : wv;
  bf16_t* outp = (which == 0) ? Qb : (which == 1) ? Kb : Vb;
  const float scale = (which == 0) ? 0.125f : 1.0f;
  x    += (size_t)(b * C_ + ct * 128) * M_;
  w    += (size_t)h * M_ * 64;
  outp += ((size_t)(b * H_ + h) * C_ + (size_t)ct * 128) * 64;

  const int tid = threadIdx.x, wave = tid >> 5, lane = tid & 31;
  const int hi = (lane >> 4), n = lane & 15;

  constexpr int XLD = 40;   // 80 B rows
  constexpr int WLD = 40;   // transposed W tile [64][40]
  __shared__ alignas(16) bf16_t XT[128 * XLD];
  __shared__ alignas(16) bf16_t WTt[64 * WLD];

  v8f acc[4] = {};

  for (int mc = 0; mc < M_ / 32; ++mc) {
#pragma unroll
    for (int i = 0; i < 4; ++i) {             // X tile 128x32 fp32, float4 loads
      int idx = tid + i * 256;
      int r = idx >> 3, c4 = (idx & 7) * 4;
      v4f v = *(const v4f*)(x + (size_t)r * M_ + mc * 32 + c4);
      v2u pk;
      pk[0] = pack2bf(v[0], v[1]);
      pk[1] = pack2bf(v[2], v[3]);
      *(v2u*)(&XT[r * XLD + c4]) = pk;
    }
#pragma unroll
    for (int i = 0; i < 2; ++i) {             // W tile 32x64 -> transposed LDS
      int idx = tid + i * 256;
      int k = idx >> 4, n4 = (idx & 15) * 4;
      v4f v = *(const v4f*)(w + (size_t)(mc * 32 + k) * 64 + n4);
#pragma unroll
      for (int j = 0; j < 4; ++j) store_bf16(&WTt[(n4 + j) * WLD + k], v[j]);
    }
    __syncthreads();
    v16bf a = load_A16(&XT[(wave * 16) * XLD], XLD, lane);
#pragma unroll
    for (int t = 0; t < 4; ++t) {
      v16bf bb = load_B16(&WTt[(t * 16) * WLD], WLD, lane);
      acc[t] = wmma_bf16(a, bb, acc[t]);
    }
    __syncthreads();
  }
#pragma unroll
  for (int t = 0; t < 4; ++t)
#pragma unroll
    for (int r = 0; r < 8; ++r) {
      int row = wave * 16 + r + 8 * hi;
      store_bf16(&outp[(size_t)row * 64 + t * 16 + n], acc[t][r] * scale);
    }
}

// ---------------------------------------------------------------------------
// Kernel 2: flash attention per (b,h).  grid = (C/128, B*H), block = 256.
// ---------------------------------------------------------------------------
__global__ __launch_bounds__(256) void flash_kernel(
    const bf16_t* __restrict__ Qb, const bf16_t* __restrict__ Kb,
    const bf16_t* __restrict__ Vb, bf16_t* __restrict__ Pre) {
  const int bh = blockIdx.y, b = bh / H_, h = bh % H_;
  const int dt = blockIdx.x;
  const int tid = threadIdx.x, wave = tid >> 5, lane = tid & 31;
  const int hi = (lane >> 4), n = lane & 15;

  constexpr int QLD = 72, KLD = 72, VLD = 72, PLD = 72;  // 144 B rows
  __shared__ alignas(16) bf16_t Qs[128 * QLD];
  __shared__ alignas(16) bf16_t Ks[64 * KLD];
  __shared__ alignas(16) bf16_t Vt[64 * VLD];            // [v][key]
  __shared__ alignas(16) bf16_t Ps[8 * 16 * PLD];        // per wave [query][key]
  bf16_t* Pw = &Ps[wave * 16 * PLD];

  const size_t base_q   = ((size_t)(b * H_ + h) * C_ + (size_t)dt * 128) * 64;
  const size_t base_kv0 = (size_t)(b * H_ + h) * C_ * 64;

#pragma unroll
  for (int i = 0; i < 4; ++i) {               // Q tile 128x64 bf16, b128 loads
    int idx = tid + i * 256;
    int r = idx >> 3, c8 = (idx & 7) * 8;
    *(v4u*)(&Qs[r * QLD + c8]) = *(const v4u*)(Qb + base_q + (size_t)r * 64 + c8);
  }

  v8f Ot[4] = {};
  float m_run = -1e30f, l_run = 0.0f;

  for (int kb = 0; kb < C_ / 64; ++kb) {
    __syncthreads();
    const size_t base_kv = base_kv0 + (size_t)kb * 64 * 64;
#pragma unroll
    for (int i = 0; i < 2; ++i) {             // K row-major, V transposed
      int idx = tid + i * 256;
      int r = idx >> 3, c8 = (idx & 7) * 8;
      *(v4u*)(&Ks[r * KLD + c8]) =
          *(const v4u*)(Kb + base_kv + (size_t)r * 64 + c8);
      v4u vv = *(const v4u*)(Vb + base_kv + (size_t)r * 64 + c8);
      v8bf ve = __builtin_bit_cast(v8bf, vv);
#pragma unroll
      for (int j = 0; j < 8; ++j) Vt[(c8 + j) * VLD + r] = ve[j];
    }
    if (kb + 1 < C_ / 64) {                   // prefetch next K/V tile
      __builtin_prefetch(Kb + base_kv + 4096 + (size_t)tid * 16, 0, 1);
      __builtin_prefetch(Vb + base_kv + 4096 + (size_t)tid * 16, 0, 1);
    }
    __syncthreads();

    // S^T: 4 key subtiles (16 keys x 16 queries), contract feature dim 64
    v8f St[4];
#pragma unroll
    for (int t = 0; t < 4; ++t) {
      v8f s = {};
#pragma unroll
      for (int kc = 0; kc < 2; ++kc) {
        v16bf a  = load_A16(&Ks[(t * 16) * KLD + kc * 32], KLD, lane);
        v16bf bq = load_B16(&Qs[(wave * 16) * QLD + kc * 32], QLD, lane);
        s = wmma_bf16(a, bq, s);
      }
      St[t] = s;
    }

    // online softmax (query = lane&15; halves combined via xor 16)
    float lmax = -1e30f;
#pragma unroll
    for (int t = 0; t < 4; ++t)
#pragma unroll
      for (int r = 0; r < 8; ++r) lmax = fmaxf(lmax, St[t][r]);
    lmax = fmaxf(lmax, __shfl_xor(lmax, 16));
    float m_new = fmaxf(m_run, lmax);
    float alpha = __expf(m_run - m_new);

    float rowsum = 0.0f;
#pragma unroll
    for (int t = 0; t < 4; ++t)
#pragma unroll
      for (int r = 0; r < 8; r += 2) {        // keys r,r+1 adjacent -> b32 store
        float p0 = __expf(St[t][r] - m_new);
        float p1 = __expf(St[t][r + 1] - m_new);
        rowsum += p0 + p1;
        int key = t * 16 + 8 * hi + r;
        *(unsigned*)(&Pw[n * PLD + key]) = pack2bf(p0, p1);
      }
    rowsum += __shfl_xor(rowsum, 16);
    l_run = alpha * l_run + rowsum;
    m_run = m_new;

#pragma unroll
    for (int r = 0; r < 8; ++r) {             // rescale O accumulators
      float scl = __shfl(alpha, r + 8 * hi);
#pragma unroll
      for (int t = 0; t < 4; ++t) Ot[t][r] *= scl;
    }

    // O += P x V  (A = P row-major, B = V transposed tile)
#pragma unroll
    for (int t = 0; t < 4; ++t)
#pragma unroll
      for (int kc = 0; kc < 2; ++kc) {
        v16bf a  = load_A16(&Pw[kc * 32], PLD, lane);
        v16bf bv = load_B16(&Vt[(t * 16) * VLD + kc * 32], VLD, lane);
        Ot[t] = wmma_bf16(a, bv, Ot[t]);
      }
  }

  float linv = 1.0f / l_run;
#pragma unroll
  for (int r = 0; r < 8; ++r) {
    float scl = __shfl(linv, r + 8 * hi);
    int d = dt * 128 + wave * 16 + r + 8 * hi;
#pragma unroll
    for (int t = 0; t < 4; ++t)
      store_bf16(&Pre[((size_t)b * C_ + d) * (H_ * 64) + h * 64 + t * 16 + n],
                 Ot[t][r] * scl);
  }
}

// ---------------------------------------------------------------------------
// Kernel 3: out[b,d,m] = pre[b,d,:1024] x wo_flat[:1024,m].
// grid = (C/128, 1024/64, B), block = 256.
// ---------------------------------------------------------------------------
__global__ __launch_bounds__(256) void out_proj_kernel(
    const bf16_t* __restrict__ Pre, const float* __restrict__ wo,
    float* __restrict__ out) {
  const int b = blockIdx.z, rt = blockIdx.x, ctile = blockIdx.y;
  const int tid = threadIdx.x, wave = tid >> 5, lane = tid & 31;
  const int hi = (lane >> 4), n = lane & 15;

  constexpr int XLD = 40, WLD = 40;
  __shared__ alignas(16) bf16_t XT[128 * XLD];
  __shared__ alignas(16) bf16_t WTt[64 * WLD];           // transposed wo tile

  v8f acc[4] = {};
  const size_t base_x = ((size_t)b * C_ + (size_t)rt * 128) * 1024;

  for (int mc = 0; mc < 1024 / 32; ++mc) {
#pragma unroll
    for (int i = 0; i < 2; ++i) {             // X tile 128x32 bf16, b128 loads
      int idx = tid + i * 256;
      int r = idx >> 2, c8 = (idx & 3) * 8;
      *(v4u*)(&XT[r * XLD + c8]) =
          *(const v4u*)(Pre + base_x + (size_t)r * 1024 + mc * 32 + c8);
    }
#pragma unroll
    for (int i = 0; i < 2; ++i) {             // wo tile 32x64 -> transposed LDS
      int idx = tid + i * 256;
      int k = idx >> 4, n4 = (idx & 15) * 4;
      v4f v = *(const v4f*)(wo + (size_t)(mc * 32 + k) * 1024 + ctile * 64 + n4);
#pragma unroll
      for (int j = 0; j < 4; ++j) store_bf16(&WTt[(n4 + j) * WLD + k], v[j]);
    }
    __syncthreads();
    v16bf a = load_A16(&XT[(wave * 16) * XLD], XLD, lane);
#pragma unroll
    for (int t = 0; t < 4; ++t) {
      v16bf bb = load_B16(&WTt[(t * 16) * WLD], WLD, lane);
      acc[t] = wmma_bf16(a, bb, acc[t]);
    }
    __syncthreads();
  }
#pragma unroll
  for (int t = 0; t < 4; ++t)
#pragma unroll
    for (int r = 0; r < 8; ++r) {
      int row = rt * 128 + wave * 16 + r + 8 * hi;
      out[((size_t)b * C_ + row) * 1024 + ctile * 64 + t * 16 + n] = acc[t][r];
    }
}

extern "C" void kernel_launch(void* const* d_in, const int* in_sizes, int n_in,
                              void* d_out, int out_size, void* d_ws,
                              size_t ws_size, hipStream_t stream) {
  const float* kvinput = (const float*)d_in[0];
  const float* qinput  = (const float*)d_in[1];
  const float* wq = (const float*)d_in[2];
  const float* wk = (const float*)d_in[3];
  const float* wv = (const float*)d_in[4];
  const float* wo = (const float*)d_in[5];
  float* out = (float*)d_out;

  const size_t qkv_elems = (size_t)B_ * H_ * C_ * 64;
  bf16_t* Qb  = (bf16_t*)d_ws;
  bf16_t* Kb  = Qb + qkv_elems;
  bf16_t* Vb  = Kb + qkv_elems;
  bf16_t* Pre = Vb + qkv_elems;

  qkv_kernel<<<dim3(C_ / 128, B_ * H_, 3), 256, 0, stream>>>(
      kvinput, qinput, wq, wk, wv, Qb, Kb, Vb);
  flash_kernel<<<dim3(C_ / 128, B_ * H_), 256, 0, stream>>>(Qb, Kb, Vb, Pre);
  out_proj_kernel<<<dim3(C_ / 128, 1024 / 64, B_), 256, 0, stream>>>(Pre, wo, out);
}
